// MultiHeadedAttention2_28544352649621
// MI455X (gfx1250) — matrix operand order repaired and checked by reading, not compiled, not run on hardware
//
#include <hip/hip_runtime.h>

// Dual-stream MHA for MI455X (gfx1250, wave32, WMMA).
// Requires ws_size >= 8 * 8*1024*512 * 4 bytes = 128 MiB.
//
// Pipeline:
//   gemm_xwt_bias x6 : Q/K/V projections (fp32 in, f16 WMMA, fp32 out) -> ws
//   dual_attn        : fused 2-pass softmax, attn = max(softmax_r, softmax_f),
//                      out_r = attn@V_r, out_f = attn@V_f            -> ws
//   gemm_xwt_bias x2 : output projections                            -> d_out

#define DEV __device__ __forceinline__

typedef __attribute__((ext_vector_type(16))) _Float16 v16h;
typedef __attribute__((ext_vector_type(8)))  _Float16 v8h;
typedef __attribute__((ext_vector_type(8)))  float    v8f;

static constexpr int    Bn = 8, Sn = 1024, Dm = 512, Hn = 8, DK = 64;
static constexpr size_t CH = (size_t)Bn * Sn * Dm;   // elements per [B,S,D] buffer

DEV v8f wmma_f16(v16h a, v16h b, v8f c) {
  return __builtin_amdgcn_wmma_f32_16x16x32_f16(false, a, false, b, (short)0, c,
                                                false, false);
}

// Load a 16-element f16 fragment as two 16B chunks from a K-major LDS row.
DEV v16h ld16(const _Float16* p, int c0, int c1) {
  v8h lo = *(const v8h*)(p + c0);
  v8h hi = *(const v8h*)(p + c1);
  v16h r;
#pragma unroll
  for (int i = 0; i < 8; ++i) { r[i] = lo[i]; r[8 + i] = hi[i]; }
  return r;
}

DEV v8f vzero() { v8f z = {0.f,0.f,0.f,0.f,0.f,0.f,0.f,0.f}; return z; }

// ---------------------------------------------------------------------------
// Generic GEMM: Y[M,N] = X[M,K] @ W[N,K]^T + bias,  M=8192, N=K=512.
// Block tile 128x128, K-step 32, 8 waves each doing a 64x32 subtile
// (4x2 v_wmma_f32_16x16x32_f16 per K-step).
// ---------------------------------------------------------------------------
__global__ __launch_bounds__(256)
void gemm_xwt_bias(const float* __restrict__ X, const float* __restrict__ W,
                   const float* __restrict__ bias, float* __restrict__ Y) {
  constexpr int LD = 40;                       // 32 + 8 halves pad: conflict-free b128
  __shared__ _Float16 Ah[128 * LD];
  __shared__ _Float16 Bh[128 * LD];

  const int m0 = blockIdx.x * 128;
  const int n0 = blockIdx.y * 128;
  const int tid  = threadIdx.x;
  const int wave = tid >> 5, lane = tid & 31;
  const int half = lane >> 4, lanen = lane & 15;
  const int wm = (wave & 1) * 64;              // wave row offset in block tile
  const int wn = (wave >> 1) * 32;             // wave col offset in block tile
  const int klo = half * 8, kb = half * 16;    // A-frag and B-frag lane chunk base (kb unused)

  v8f acc[4][2];
#pragma unroll
  for (int i = 0; i < 4; ++i)
#pragma unroll
    for (int j = 0; j < 2; ++j) acc[i][j] = vzero();

  for (int kt = 0; kt < 512; kt += 32) {
    // Stage X tile [128][32] and W tile [128][32], fp32 -> f16, K-major.
#pragma unroll
    for (int it = 0; it < 4; ++it) {
      int idx = tid + 256 * it;                // 1024 float4 chunks per tile
      int row = idx >> 3, c = (idx & 7) << 2;
      float4 a = *(const float4*)(X + (size_t)(m0 + row) * 512 + kt + c);
      _Float16* d = &Ah[row * LD + c];
      d[0]=(_Float16)a.x; d[1]=(_Float16)a.y; d[2]=(_Float16)a.z; d[3]=(_Float16)a.w;
      float4 b = *(const float4*)(W + (size_t)(n0 + row) * 512 + kt + c);
      _Float16* e = &Bh[row * LD + c];
      e[0]=(_Float16)b.x; e[1]=(_Float16)b.y; e[2]=(_Float16)b.z; e[3]=(_Float16)b.w;
    }
    __syncthreads();

    v16h bfr[2];
#pragma unroll
    for (int j = 0; j < 2; ++j)
      bfr[j] = ld16(&Bh[(wn + j * 16 + lanen) * LD], klo, klo + 16);
#pragma unroll
    for (int i = 0; i < 4; ++i) {
      v16h af = ld16(&Ah[(wm + i * 16 + lanen) * LD], klo, klo + 16);
#pragma unroll
      for (int j = 0; j < 2; ++j) acc[i][j] = wmma_f16(af, bfr[j], acc[i][j]);
    }
    __syncthreads();
  }

  // Epilogue: C/D layout = VGPR r -> row (r + 8*half), lane -> col.
#pragma unroll
  for (int j = 0; j < 2; ++j) {
    int col = n0 + wn + j * 16 + lanen;
    float bv = bias[col];
#pragma unroll
    for (int i = 0; i < 4; ++i) {
      int rbase = m0 + wm + i * 16 + half * 8;
#pragma unroll
      for (int r = 0; r < 8; ++r)
        Y[(size_t)(rbase + r) * 512 + col] = acc[i][j][r] + bv;
    }
  }
}

// ---------------------------------------------------------------------------
// Fused dual-stream attention. One block = one (b,h) pair + 128 q-rows.
// 8 waves; each wave privately owns 16 q-rows (no cross-wave reductions).
// Pass 1: online (max, sum) for both streams. Pass 2: recompute scores
// (K/V are L2-resident), p = max(p_rgb, p_flow), accumulate p@V_r, p@V_f.
// ---------------------------------------------------------------------------
__global__ __launch_bounds__(256)
void dual_attn(const float* __restrict__ Qr, const float* __restrict__ Kr,
               const float* __restrict__ Vr, const float* __restrict__ Qf,
               const float* __restrict__ Kf, const float* __restrict__ Vf,
               float* __restrict__ Or, float* __restrict__ Of) {
  constexpr int LK = 72;  // K tile [32 keys][64 d] stride (64+8 pad)
  constexpr int LV = 40;  // V^T tile [64 d][32 keys] stride
  constexpr int LP = 40;  // P tile [16 q][32 keys] stride
  __shared__ _Float16 Khr[32 * LK], Khf[32 * LK];
  __shared__ _Float16 Vtr[64 * LV], Vtf[64 * LV];
  __shared__ _Float16 Pt[8][16 * LP];          // per-wave probability tile

  const int qt = blockIdx.x & 7;               // 8 q-tiles of 128 rows
  const int bh = blockIdx.x >> 3;
  const int b = bh >> 3, h = bh & 7;
  const int tid = threadIdx.x, wave = tid >> 5, lane = tid & 31;
  const int half = lane >> 4, lanen = lane & 15;
  const int klo = half * 8, kb = half * 16;
  const float scale = 0.125f;                  // 1/sqrt(64)

  const size_t headOff = (size_t)b * Sn * Dm + h * DK;
  const int qbase = qt * 128 + wave * 16;

  // ---- Q fragments: [16 q][64 d] = 2 A-frags per stream, loaded once ----
  v16h qAr[2], qAf[2];
  {
    const float* rr = Qr + headOff + (size_t)(qbase + lanen) * Dm;
    const float* rf = Qf + headOff + (size_t)(qbase + lanen) * Dm;
#pragma unroll
    for (int f = 0; f < 2; ++f) {
      float t[16];
#pragma unroll
      for (int g = 0; g < 2; ++g) {
        float4 x0 = *(const float4*)(rr + 32 * f + klo + 16 * g);
        float4 x1 = *(const float4*)(rr + 32 * f + klo + 16 * g + 4);
        t[8*g+0]=x0.x; t[8*g+1]=x0.y; t[8*g+2]=x0.z; t[8*g+3]=x0.w;
        t[8*g+4]=x1.x; t[8*g+5]=x1.y; t[8*g+6]=x1.z; t[8*g+7]=x1.w;
      }
#pragma unroll
      for (int i = 0; i < 16; ++i) qAr[f][i] = (_Float16)t[i];
#pragma unroll
      for (int g = 0; g < 2; ++g) {
        float4 x0 = *(const float4*)(rf + 32 * f + klo + 16 * g);
        float4 x1 = *(const float4*)(rf + 32 * f + klo + 16 * g + 4);
        t[8*g+0]=x0.x; t[8*g+1]=x0.y; t[8*g+2]=x0.z; t[8*g+3]=x0.w;
        t[8*g+4]=x1.x; t[8*g+5]=x1.y; t[8*g+6]=x1.z; t[8*g+7]=x1.w;
      }
#pragma unroll
      for (int i = 0; i < 16; ++i) qAf[f][i] = (_Float16)t[i];
    }
  }

  // ---- staging helpers (cooperative, 256 threads) ----
  auto stage_k = [&](_Float16* dst, const float* src) {
#pragma unroll
    for (int it = 0; it < 2; ++it) {
      int idx = tid + 256 * it;                // 512 float4 over [32][64]
      int row = idx >> 4, c = (idx & 15) << 2;
      float4 v = *(const float4*)(src + (size_t)row * Dm + c);
      _Float16* d = dst + row * LK + c;
      d[0]=(_Float16)v.x; d[1]=(_Float16)v.y; d[2]=(_Float16)v.z; d[3]=(_Float16)v.w;
    }
  };
  auto stage_vt = [&](_Float16* dst, const float* src) {
#pragma unroll
    for (int it = 0; it < 2; ++it) {
      int idx = tid + 256 * it;
      int row = idx >> 4, c = (idx & 15) << 2; // row = key, c = d
      float4 v = *(const float4*)(src + (size_t)row * Dm + c);
      dst[(c + 0) * LV + row] = (_Float16)v.x;
      dst[(c + 1) * LV + row] = (_Float16)v.y;
      dst[(c + 2) * LV + row] = (_Float16)v.z;
      dst[(c + 3) * LV + row] = (_Float16)v.w;
    }
  };
  // scores for one stream over a 32-key tile: two 16x16 C tiles
  auto scores2 = [&](const v16h& a0, const v16h& a1, const _Float16* Kt,
                     v8f& s0, v8f& s1) {
    v16h b0 = ld16(Kt + (lanen) * LK,       klo,      klo + 16);   // keys 0-15, d 0-31
    v16h b1 = ld16(Kt + (lanen) * LK,       32 + klo, 32 + klo + 16);
    s0 = wmma_f16(a1, b1, wmma_f16(a0, b0, vzero()));
    v16h c0 = ld16(Kt + (16 + lanen) * LK,  klo,      klo + 16);   // keys 16-31
    v16h c1 = ld16(Kt + (16 + lanen) * LK,  32 + klo, 32 + klo + 16);
    s1 = wmma_f16(a1, c1, wmma_f16(a0, c0, vzero()));
#pragma unroll
    for (int r = 0; r < 8; ++r) { s0[r] *= scale; s1[r] *= scale; }
  };
  // online softmax stats; row r of the C tile lives in a 16-lane half
  auto upd = [&](const v8f& s0, const v8f& s1, float m[8], float z[8]) {
#pragma unroll
    for (int r = 0; r < 8; ++r) {
      float vmax = fmaxf(s0[r], s1[r]);
#pragma unroll
      for (int off = 1; off < 16; off <<= 1)
        vmax = fmaxf(vmax, __shfl_xor(vmax, off, 32));
      float mn = fmaxf(m[r], vmax);
      float e = __expf(s0[r] - mn) + __expf(s1[r] - mn);
#pragma unroll
      for (int off = 1; off < 16; off <<= 1) e += __shfl_xor(e, off, 32);
      z[r] = z[r] * __expf(m[r] - mn) + e;
      m[r] = mn;
    }
  };

  float mR[8], zR[8], mF[8], zF[8];
#pragma unroll
  for (int r = 0; r < 8; ++r) { mR[r] = -1e30f; zR[r] = 0.f; mF[r] = -1e30f; zF[r] = 0.f; }

  // ---------------- Pass 1: row max & sum for both streams ----------------
  for (int kt = 0; kt < Sn / 32; ++kt) {
    const float* kr = Kr + headOff + (size_t)kt * 32 * Dm;
    const float* kf = Kf + headOff + (size_t)kt * 32 * Dm;
    stage_k(Khr, kr);
    stage_k(Khf, kf);
    if (kt + 1 < Sn / 32) {                    // global_prefetch_b8 next tiles
      __builtin_prefetch(kr + 32 * Dm + tid * 8, 0, 1);
      __builtin_prefetch(kf + 32 * Dm + tid * 8, 0, 1);
    }
    __syncthreads();
    v8f s0, s1, t0, t1;
    scores2(qAr[0], qAr[1], Khr, s0, s1);
    scores2(qAf[0], qAf[1], Khf, t0, t1);
    upd(s0, s1, mR, zR);
    upd(t0, t1, mF, zF);
    __syncthreads();
  }
  float izR[8], izF[8];
#pragma unroll
  for (int r = 0; r < 8; ++r) { izR[r] = 1.f / zR[r]; izF[r] = 1.f / zF[r]; }

  // ---------------- Pass 2: merged probs, accumulate P@V ------------------
  v8f oR[4], oF[4];
#pragma unroll
  for (int j = 0; j < 4; ++j) { oR[j] = vzero(); oF[j] = vzero(); }

  for (int kt = 0; kt < Sn / 32; ++kt) {
    const float* kr = Kr + headOff + (size_t)kt * 32 * Dm;
    const float* kf = Kf + headOff + (size_t)kt * 32 * Dm;
    const float* vr = Vr + headOff + (size_t)kt * 32 * Dm;
    const float* vf = Vf + headOff + (size_t)kt * 32 * Dm;
    stage_k(Khr, kr);
    stage_k(Khf, kf);
    stage_vt(Vtr, vr);
    stage_vt(Vtf, vf);
    if (kt + 1 < Sn / 32) {
      __builtin_prefetch(vr + 32 * Dm + tid * 8, 0, 1);
      __builtin_prefetch(vf + 32 * Dm + tid * 8, 0, 1);
    }
    __syncthreads();

    v8f s0, s1, t0, t1;
    scores2(qAr[0], qAr[1], Khr, s0, s1);
    scores2(qAf[0], qAf[1], Khf, t0, t1);

    _Float16* Pw = Pt[wave];                   // wave-private; no block barrier
#pragma unroll
    for (int r = 0; r < 8; ++r) {
      float p0 = fmaxf(__expf(s0[r] - mR[r]) * izR[r],
                       __expf(t0[r] - mF[r]) * izF[r]);
      float p1 = fmaxf(__expf(s1[r] - mR[r]) * izR[r],
                       __expf(t1[r] - mF[r]) * izF[r]);
      Pw[(r + 8 * half) * LP + lanen]      = (_Float16)p0;
      Pw[(r + 8 * half) * LP + 16 + lanen] = (_Float16)p1;
    }
    asm volatile("s_wait_dscnt 0x0" ::: "memory");  // wave-local LDS RAW fence

    v16h pa = ld16(Pw + lanen * LP, klo, klo + 16); // C-layout -> A-layout via LDS
#pragma unroll
    for (int jb = 0; jb < 4; ++jb) {
      v16h bv = ld16(&Vtr[(jb * 16 + lanen) * LV], klo, klo + 16);
      oR[jb] = wmma_f16(pa, bv, oR[jb]);
      v16h bw = ld16(&Vtf[(jb * 16 + lanen) * LV], klo, klo + 16);
      oF[jb] = wmma_f16(pa, bw, oF[jb]);
    }
    __syncthreads();
  }

  // ---- write per-head outputs back into [B,S,D] layout ----
#pragma unroll
  for (int jb = 0; jb < 4; ++jb) {
    int d = jb * 16 + lanen;
#pragma unroll
    for (int r = 0; r < 8; ++r) {
      int row = qbase + r + 8 * half;
      Or[headOff + (size_t)row * Dm + d] = oR[jb][r];
      Of[headOff + (size_t)row * Dm + d] = oF[jb][r];
    }
  }
}

// ---------------------------------------------------------------------------
extern "C" void kernel_launch(void* const* d_in, const int* in_sizes, int n_in,
                              void* d_out, int out_size, void* d_ws, size_t ws_size,
                              hipStream_t stream) {
  const float* x_rgb  = (const float*)d_in[0];
  const float* x_flow = (const float*)d_in[1];
  const float* Wq1 = (const float*)d_in[2];
  const float* Wk1 = (const float*)d_in[3];
  const float* Wv1 = (const float*)d_in[4];
  const float* Wq2 = (const float*)d_in[5];
  const float* Wk2 = (const float*)d_in[6];
  const float* Wv2 = (const float*)d_in[7];
  const float* Wo1 = (const float*)d_in[8];
  const float* Wo2 = (const float*)d_in[9];
  const float* bq1 = (const float*)d_in[10];
  const float* bk1 = (const float*)d_in[11];
  const float* bv1 = (const float*)d_in[12];
  const float* bq2 = (const float*)d_in[13];
  const float* bk2 = (const float*)d_in[14];
  const float* bv2 = (const float*)d_in[15];
  const float* bo1 = (const float*)d_in[16];
  const float* bo2 = (const float*)d_in[17];

  float* ws = (float*)d_ws;                 // needs 8*CH floats = 128 MiB
  float* Qr = ws + 0 * CH;
  float* Kr = ws + 1 * CH;
  float* Vr = ws + 2 * CH;
  float* Qf = ws + 3 * CH;
  float* Kf = ws + 4 * CH;
  float* Vf = ws + 5 * CH;
  float* Ar = ws + 6 * CH;                  // attention outputs (pre-projection)
  float* Af = ws + 7 * CH;

  float* out0 = (float*)d_out;              // projected rgb output
  float* out1 = (float*)d_out + CH;         // projected flow output

  dim3 gridG(64, 4);                        // M/128 x N/128
  gemm_xwt_bias<<<gridG, 256, 0, stream>>>(x_rgb,  Wq1, bq1, Qr);
  gemm_xwt_bias<<<gridG, 256, 0, stream>>>(x_rgb,  Wk1, bk1, Kr);
  gemm_xwt_bias<<<gridG, 256, 0, stream>>>(x_rgb,  Wv1, bv1, Vr);
  gemm_xwt_bias<<<gridG, 256, 0, stream>>>(x_flow, Wq2, bq2, Qf);
  gemm_xwt_bias<<<gridG, 256, 0, stream>>>(x_flow, Wk2, bk2, Kf);
  gemm_xwt_bias<<<gridG, 256, 0, stream>>>(x_flow, Wv2, bv2, Vf);

  dual_attn<<<dim3(Bn * Hn * (Sn / 128)), 256, 0, stream>>>(Qr, Kr, Vr, Qf, Kf, Vf, Ar, Af);

  gemm_xwt_bias<<<gridG, 256, 0, stream>>>(Ar, Wo1, bo1, out0);
  gemm_xwt_bias<<<gridG, 256, 0, stream>>>(Af, Wo2, bo2, out1);
}
